// Block_609885356912
// MI455X (gfx1250) — matrix operand order repaired and checked by reading, not compiled, not run on hardware
//
#include <hip/hip_runtime.h>
#include <cmath>

typedef __bf16 bf16;
typedef __attribute__((ext_vector_type(16))) __bf16 v16bf;
typedef __attribute__((ext_vector_type(8)))  float  v8f;

union FragB { v16bf v; unsigned u[8]; };

#define WS_   14
#define NH_   12
#define HD_   64
#define C_    768
#define C3_   2304
#define MLP_  3072
#define BDIM  8
#define HDIM  64
#define WDIM  64
#define NWIN  200          // 8 * 5 * 5 windows
#define SQ    196          // 14*14 tokens per window
#define NTOKW (NWIN * SQ)  // 39200 window tokens
#define NTOK  (BDIM * HDIM * WDIM) // 32768 image tokens
#define KTPAD 224          // key dim padded to 7*32 for attention loop
#define NWH   (NWIN * NH_) // 2400 (window, head) pairs

// A-fragment (16x32 bf16) K index for VGPR j, lane-half h (ISA 7.12.2)
__device__ __forceinline__ int a_kk(int j, int half) {
  return (j < 4) ? (half * 8 + 2 * j) : (16 + half * 8 + 2 * (j - 4));
}

// Async global->LDS 16B copy (ASYNCcnt path). VDST operand = LDS byte offset
// (low 32 bits of the generic address per the LDS aperture mapping).
__device__ __forceinline__ void async_copy_b128(void* lds_ptr,
                                                const void* gptr) {
  unsigned lds_off = (unsigned)(unsigned long long)lds_ptr;
  asm volatile("global_load_async_to_lds_b128 %0, %1, off"
               :: "v"(lds_off), "v"((unsigned long long)gptr)
               : "memory");
}
__device__ __forceinline__ void wait_async0() {
  asm volatile("s_wait_asynccnt 0x0" ::: "memory");
}

// ---------------------------------------------------------------------------
// Weight prep: qkv_w + lora_B@lora_A folded, everything cast to bf16
// ---------------------------------------------------------------------------
__global__ __launch_bounds__(256) void fold_weights_kernel(
    const float* __restrict__ qkv_w, const float* __restrict__ lora_A,
    const float* __restrict__ lora_B, const float* __restrict__ proj_w,
    const float* __restrict__ mlp_w1, const float* __restrict__ mlp_w2,
    bf16* __restrict__ wqkv, bf16* __restrict__ wproj,
    bf16* __restrict__ wm1, bf16* __restrict__ wm2) {
  long i = (long)blockIdx.x * 256 + threadIdx.x;
  const long nq = (long)C3_ * C_;
  const long np = (long)C_ * C_;
  const long n1 = (long)MLP_ * C_;
  const long n2 = (long)C_ * MLP_;
  if (i < nq) {
    int n = (int)(i / C_), k = (int)(i - (long)n * C_);
    float acc = qkv_w[i];
#pragma unroll
    for (int r = 0; r < 12; ++r) acc += lora_B[n * 12 + r] * lora_A[r * C_ + k];
    wqkv[i] = (bf16)acc;
  } else if (i < nq + np) {
    wproj[i - nq] = (bf16)proj_w[i - nq];
  } else if (i < nq + np + n1) {
    wm1[i - nq - np] = (bf16)mlp_w1[i - nq - np];
  } else if (i < nq + np + n1 + n2) {
    wm2[i - nq - np - n1] = (bf16)mlp_w2[i - nq - np - n1];
  }
}

// ---------------------------------------------------------------------------
// LN1 + pad + window partition: x[B,H,W,C] -> xw bf16 [39200, 768]
// ---------------------------------------------------------------------------
__global__ __launch_bounds__(256) void ln1_window_kernel(
    const float* __restrict__ x, const float* __restrict__ w,
    const float* __restrict__ b, bf16* __restrict__ xw) {
  const int row = blockIdx.x, tid = threadIdx.x;
  int win = row / SQ, tok = row - win * SQ;
  int bb = win / 25, wi = win - bb * 25;
  int nh = wi / 5, nw = wi - nh * 5;
  int hh = nh * WS_ + tok / WS_, ww = nw * WS_ + tok % WS_;
  if (hh >= HDIM || ww >= WDIM) {  // spatial padding: zeros (pad is after LN)
    for (int c = tid; c < C_; c += 256) xw[(long)row * C_ + c] = (bf16)0.0f;
    return;
  }
  const float* xr = x + (((long)(bb * HDIM + hh)) * WDIM + ww) * C_;
  float s = 0.f, s2 = 0.f;
  for (int c = tid; c < C_; c += 256) { float v = xr[c]; s += v; s2 += v * v; }
#pragma unroll
  for (int off = 16; off >= 1; off >>= 1) {
    s += __shfl_xor(s, off, 32); s2 += __shfl_xor(s2, off, 32);
  }
  __shared__ float red[2][8];
  if ((tid & 31) == 0) { red[0][tid >> 5] = s; red[1][tid >> 5] = s2; }
  __syncthreads();
  float ts = 0.f, t2 = 0.f;
#pragma unroll
  for (int i = 0; i < 8; ++i) { ts += red[0][i]; t2 += red[1][i]; }
  float mean = ts * (1.0f / C_);
  float var = t2 * (1.0f / C_) - mean * mean;
  float rstd = rsqrtf(var + 1e-5f);
  for (int c = tid; c < C_; c += 256)
    xw[(long)row * C_ + c] = (bf16)((xr[c] - mean) * rstd * w[c] + b[c]);
}

// ---------------------------------------------------------------------------
// LN2 over d_out f32 [32768,768] -> h bf16
// ---------------------------------------------------------------------------
__global__ __launch_bounds__(256) void ln2_kernel(
    const float* __restrict__ y, const float* __restrict__ w,
    const float* __restrict__ b, bf16* __restrict__ h) {
  const int row = blockIdx.x, tid = threadIdx.x;
  const float* xr = y + (long)row * C_;
  float s = 0.f, s2 = 0.f;
  for (int c = tid; c < C_; c += 256) { float v = xr[c]; s += v; s2 += v * v; }
#pragma unroll
  for (int off = 16; off >= 1; off >>= 1) {
    s += __shfl_xor(s, off, 32); s2 += __shfl_xor(s2, off, 32);
  }
  __shared__ float red[2][8];
  if ((tid & 31) == 0) { red[0][tid >> 5] = s; red[1][tid >> 5] = s2; }
  __syncthreads();
  float ts = 0.f, t2 = 0.f;
#pragma unroll
  for (int i = 0; i < 8; ++i) { ts += red[0][i]; t2 += red[1][i]; }
  float mean = ts * (1.0f / C_);
  float var = t2 * (1.0f / C_) - mean * mean;
  float rstd = rsqrtf(var + 1e-5f);
  for (int c = tid; c < C_; c += 256)
    h[(long)row * C_ + c] = (bf16)((xr[c] - mean) * rstd * w[c] + b[c]);
}

// ---------------------------------------------------------------------------
// Generic bf16 WMMA GEMM: C[M,N] = A[M,K] * W[N,K]^T, fused epilogues.
// Block tile 64x128, 8 waves -> 32x32 per wave (2x2 of 16x16), K-step 32.
// A tile staged via GLOBAL_LOAD_ASYNC_TO_LDS_B128 (ASYNCcnt).
// ---------------------------------------------------------------------------
struct GemmParams {
  const bf16* A; const bf16* W;
  int M, N, K, mode;
  const float* bias;
  bf16* q; bf16* kT; bf16* v;      // mode 0: qkv scatter
  const float* xres; float* outf;  // mode 1 (residual) / mode 3 (accumulate)
  bf16* gout;                      // mode 2: GELU output
};

__device__ __forceinline__ void gemm_epilogue(const GemmParams& p, int row,
                                              int col, float acc) {
  if (row >= p.M) return;
  if (p.mode == 0) {  // qkv: bias + split/scatter, kT stored d-major padded
    float val = acc + p.bias[col];
    int qi = col / C_;
    int rem = col - qi * C_;
    int hd = rem >> 6, d = rem & 63;
    int win = row / SQ, tok = row - win * SQ;
    long wh = (long)win * NH_ + hd;
    if (qi == 0)      p.q[(wh * SQ + tok) * HD_ + d] = (bf16)val;
    else if (qi == 1) p.kT[(wh * HD_ + d) * KTPAD + tok] = (bf16)val;
    else              p.v[(wh * SQ + tok) * HD_ + d] = (bf16)val;
  } else if (p.mode == 1) {  // proj: window-unpartition + crop + residual
    int win = row / SQ, tok = row - win * SQ;
    int bb = win / 25, wi = win - bb * 25;
    int nh = wi / 5, nw = wi - nh * 5;
    int hh = nh * WS_ + tok / WS_, ww = nw * WS_ + tok % WS_;
    if (hh < HDIM && ww < WDIM) {
      long o = (((long)(bb * HDIM + hh)) * WDIM + ww) * C_ + col;
      p.outf[o] = p.xres[o] + acc + p.bias[col];
    }
  } else if (p.mode == 2) {  // mlp1: bias + exact GELU -> bf16
    float xv = acc + p.bias[col];
    float g = 0.5f * xv * (1.0f + erff(xv * 0.7071067811865476f));
    p.gout[(long)row * p.N + col] = (bf16)g;
  } else {  // mlp2: accumulate into d_out
    long o = (long)row * C_ + col;
    p.outf[o] += acc + p.bias[col];
  }
}

__global__ __launch_bounds__(256) void gemm_bf16_kernel(GemmParams p) {
  __shared__ bf16 As[64][40];    // A tile [m][k], row stride 80B (16B aligned)
  __shared__ bf16 Bs[32][136];   // W tile transposed: Bs[k][n]
  const int tid = threadIdx.x;
  const int wave = tid >> 5, lane = tid & 31;
  const int half = lane >> 4, mlane = lane & 15;
  const int wm = wave & 1, wn = wave >> 1;
  const int blockM = blockIdx.y * 64;
  const int blockN = blockIdx.x * 128;

  v8f acc[2][2] = {};
  const int kSteps = p.K >> 5;
  for (int kt = 0; kt < kSteps; ++kt) {
    const int k0 = kt << 5;
    {  // stage A: one async 16B DMA per thread, zero-fill M tail
      int r = tid >> 2, kq = (tid & 3) << 3;
      int m = blockM + r;
      if (m < p.M) {
        async_copy_b128(&As[r][kq], p.A + (long)m * p.K + k0 + kq);
      } else {
        uint4 z = {0u, 0u, 0u, 0u};
        *(uint4*)&As[r][kq] = z;
      }
    }
#pragma unroll
    for (int it = 0; it < 2; ++it) {  // stage W transposed (register path)
      int idx = tid + (it << 8);
      int n = idx >> 2, kq = (idx & 3) << 3;
      int gn = blockN + n;  // N is always a multiple of 128 here
      uint4 t = *(const uint4*)(p.W + (long)gn * p.K + k0 + kq);
      const bf16* tb = (const bf16*)&t;
#pragma unroll
      for (int i = 0; i < 8; ++i) Bs[kq + i][n] = tb[i];
    }
    if (kt + 1 < kSteps) {  // prefetch next tiles (global_prefetch_b8)
      __builtin_prefetch(p.A + (long)(blockM + (tid >> 2)) * p.K + k0 + 32, 0, 1);
      __builtin_prefetch(p.W + (long)(blockN + (tid >> 1)) * p.K + k0 + 32, 0, 1);
    }
    wait_async0();
    __syncthreads();

    FragB a[2], b[2];
#pragma unroll
    for (int s = 0; s < 2; ++s) {
      int ar = wm * 32 + s * 16 + mlane;
#pragma unroll
      for (int j = 0; j < 8; ++j)
        a[s].u[j] = *(const unsigned*)&As[ar][a_kk(j, half)];
    }
#pragma unroll
    for (int s = 0; s < 2; ++s) {
      int nb = wn * 32 + s * 16;
#pragma unroll
      for (int j = 0; j < 8; ++j)
      {
        b[s].v[2 * j] = Bs[a_kk(j, half)][nb + mlane];
        b[s].v[2 * j + 1] = Bs[a_kk(j, half) + 1][nb + mlane];
      }
    }
#pragma unroll
    for (int s = 0; s < 2; ++s)
#pragma unroll
      for (int s2 = 0; s2 < 2; ++s2)
        acc[s][s2] = __builtin_amdgcn_wmma_f32_16x16x32_bf16(
            false, a[s].v, false, b[s2].v, (short)0, acc[s][s2], false, false);
    __syncthreads();
  }
#pragma unroll
  for (int s = 0; s < 2; ++s)
#pragma unroll
    for (int s2 = 0; s2 < 2; ++s2)
#pragma unroll
      for (int j = 0; j < 8; ++j) {
        int row = blockM + wm * 32 + s * 16 + j + half * 8;
        int col = blockN + wn * 32 + s2 * 16 + mlane;
        gemm_epilogue(p, row, col, acc[s][s2][j]);
      }
}

// ---------------------------------------------------------------------------
// Flash attention per (window, head): S=196 queries/keys, D=64.
// Online softmax, 32-key steps, decomposed rel-pos bias via 196x27 dot tables.
// V tile staged with async global->LDS DMA.
// ---------------------------------------------------------------------------
__global__ __launch_bounds__(256) void attention_kernel(
    const bf16* __restrict__ q, const bf16* __restrict__ kT,
    const bf16* __restrict__ v, const float* __restrict__ rel_h,
    const float* __restrict__ rel_w, bf16* __restrict__ attn_out) {
  __shared__ bf16 Vs[KTPAD][HD_];   // 28672 B
  __shared__ bf16 Th[SQ][28];       // 10976 B  (q . rel_pos_h[idx])
  __shared__ bf16 Tw[SQ][28];       // 10976 B
  __shared__ bf16 Pst[8][16][32];   //  8192 B  P-tile C->A relayout per wave

  const int tid = threadIdx.x;
  const int wave = tid >> 5, lane = tid & 31;
  const int half = lane >> 4, mlane = lane & 15;
  const long wh = blockIdx.x;
  const int win = (int)(wh / NH_), head = (int)(wh - (long)win * NH_);

  const bf16* qb = q + wh * SQ * HD_;
  const bf16* kb_ = kT + wh * HD_ * KTPAD;
  const bf16* vb = v + wh * SQ * HD_;

  // stage V via async DMA (16B chunks), zero-fill pad rows 196..223
  for (int c = tid; c < (KTPAD * HD_) / 8; c += 256) {
    int r = c >> 3;  // 8 chunks per 64-element row
    char* ldst = (char*)&Vs[0][0] + c * 16;
    if (r < SQ) {
      async_copy_b128(ldst, vb + c * 8);
    } else {
      uint4 z = {0u, 0u, 0u, 0u};
      *(uint4*)ldst = z;
    }
  }
  for (int idx = tid; idx < SQ * 27; idx += 256) {      // rel-pos dot tables
    int m = idx / 27, i = idx - m * 27;
    float ah = 0.f, aw = 0.f;
    const bf16* qr = qb + m * HD_;
#pragma unroll 8
    for (int d = 0; d < HD_; ++d) {
      float qv = (float)qr[d];
      ah += qv * rel_h[i * HD_ + d];
      aw += qv * rel_w[i * HD_ + d];
    }
    Th[m][i] = (bf16)ah;
    Tw[m][i] = (bf16)aw;
  }
  wait_async0();
  __syncthreads();

  for (int qt = wave; qt < 13; qt += 8) {  // 13 q-tiles of 16 rows
    const int m0 = qt * 16;
    FragB aq0, aq1;
    {
      int m = m0 + mlane;
      if (m < SQ) {
        const bf16* qr = qb + m * HD_;
#pragma unroll
        for (int j = 0; j < 8; ++j) {
          int kk = a_kk(j, half);
          aq0.u[j] = *(const unsigned*)&qr[kk];
          aq1.u[j] = *(const unsigned*)&qr[kk + 32];
        }
      } else {
#pragma unroll
        for (int j = 0; j < 8; ++j) { aq0.u[j] = 0u; aq1.u[j] = 0u; }
      }
    }
    float mrow[8], lrow[8];
#pragma unroll
    for (int j = 0; j < 8; ++j) { mrow[j] = -1e30f; lrow[j] = 0.f; }
    v8f o[4] = {};

    for (int ks = 0; ks < 7; ++ks) {  // 224 keys in steps of 32
      const int kb0 = ks * 32;
      v8f sc[2];
#pragma unroll
      for (int t = 0; t < 2; ++t) {   // two 16-key score tiles
        const int n0 = kb0 + t * 16;
        FragB blo, bhi;
#pragma unroll
        for (int j = 0; j < 8; ++j) {
          int kk = a_kk(j, half);
          blo.v[2 * j] = kb_[kk * KTPAD + n0 + mlane];
          blo.v[2 * j + 1] = kb_[(kk + 1) * KTPAD + n0 + mlane];
          bhi.v[2 * j] = kb_[(32 + kk) * KTPAD + n0 + mlane];
          bhi.v[2 * j + 1] = kb_[(33 + kk) * KTPAD + n0 + mlane];
        }
        v8f z = {};
        v8f c0 = __builtin_amdgcn_wmma_f32_16x16x32_bf16(
            false, aq0.v, false, blo.v, (short)0, z, false, false);
        sc[t] = __builtin_amdgcn_wmma_f32_16x16x32_bf16(
            false, aq1.v, false, bhi.v, (short)0, c0, false, false);
      }
      // scale + rel-pos bias + key mask
#pragma unroll
      for (int j = 0; j < 8; ++j) {
        int row = m0 + j + half * 8;
        int rr = (row < SQ) ? row : 0;
        int qh = rr / WS_, qw = rr - qh * WS_;
#pragma unroll
        for (int t = 0; t < 2; ++t) {
          int key = kb0 + t * 16 + mlane;
          float s;
          if (key < SQ) {
            int kh = key / WS_, kw = key - kh * WS_;
            s = sc[t][j] * 0.125f + (float)Th[rr][qh - kh + 13] +
                (float)Tw[rr][qw - kw + 13];
          } else {
            s = -1e30f;
          }
          sc[t][j] = s;
        }
      }
      // online softmax: per-row reduce across the 16-lane group
#pragma unroll
      for (int j = 0; j < 8; ++j) {
        float smax = fmaxf(sc[0][j], sc[1][j]);
#pragma unroll
        for (int off = 8; off >= 1; off >>= 1)
          smax = fmaxf(smax, __shfl_xor(smax, off, 32));
        float mnew = fmaxf(mrow[j], smax);
        float p0 = __expf(sc[0][j] - mnew);
        float p1 = __expf(sc[1][j] - mnew);
        float ps = p0 + p1;
#pragma unroll
        for (int off = 8; off >= 1; off >>= 1)
          ps += __shfl_xor(ps, off, 32);
        float alpha = __expf(mrow[j] - mnew);
        lrow[j] = lrow[j] * alpha + ps;
        mrow[j] = mnew;
#pragma unroll
        for (int nt = 0; nt < 4; ++nt) o[nt][j] *= alpha;
        int rl = j + half * 8;
        Pst[wave][rl][mlane] = (bf16)p0;
        Pst[wave][rl][16 + mlane] = (bf16)p1;
      }
      asm volatile("s_wait_dscnt 0x0" ::: "memory");  // cross-lane LDS RAW
      FragB ap;
#pragma unroll
      for (int j = 0; j < 8; ++j)
        ap.u[j] = *(const unsigned*)&Pst[wave][mlane][a_kk(j, half)];
#pragma unroll
      for (int nt = 0; nt < 4; ++nt) {
        FragB bv;
#pragma unroll
        for (int j = 0; j < 8; ++j)
        {
          bv.v[2 * j] = Vs[kb0 + a_kk(j, half)][nt * 16 + mlane];
          bv.v[2 * j + 1] = Vs[kb0 + a_kk(j, half) + 1][nt * 16 + mlane];
        }
        o[nt] = __builtin_amdgcn_wmma_f32_16x16x32_bf16(
            false, ap.v, false, bv.v, (short)0, o[nt], false, false);
      }
    }
    // normalize and store: attn_out layout [win*196+tok][head*64+d]
#pragma unroll
    for (int j = 0; j < 8; ++j) {
      int row = m0 + j + half * 8;
      if (row < SQ) {
        float rinv = 1.0f / lrow[j];
        long base = ((long)win * SQ + row) * C_ + head * HD_;
#pragma unroll
        for (int nt = 0; nt < 4; ++nt)
          attn_out[base + nt * 16 + mlane] = (bf16)(o[nt][j] * rinv);
      }
    }
  }
}

// ---------------------------------------------------------------------------
extern "C" void kernel_launch(void* const* d_in, const int* in_sizes, int n_in,
                              void* d_out, int out_size, void* d_ws,
                              size_t ws_size, hipStream_t stream) {
  const float* x       = (const float*)d_in[0];
  const float* norm1_w = (const float*)d_in[1];
  const float* norm1_b = (const float*)d_in[2];
  const float* qkv_w   = (const float*)d_in[3];
  const float* qkv_b   = (const float*)d_in[4];
  const float* lora_A  = (const float*)d_in[5];
  const float* lora_B  = (const float*)d_in[6];
  const float* proj_w  = (const float*)d_in[7];
  const float* proj_b  = (const float*)d_in[8];
  const float* rel_h   = (const float*)d_in[9];
  const float* rel_w   = (const float*)d_in[10];
  const float* norm2_w = (const float*)d_in[11];
  const float* norm2_b = (const float*)d_in[12];
  const float* mlp_w1  = (const float*)d_in[13];
  const float* mlp_b1  = (const float*)d_in[14];
  const float* mlp_w2  = (const float*)d_in[15];
  const float* mlp_b2  = (const float*)d_in[16];
  float* out = (float*)d_out;

  char* ws = (char*)d_ws;
  size_t off = 0;
  auto alloc = [&](size_t bytes) -> char* {
    char* p = ws + off;
    off += (bytes + 255) & ~(size_t)255;
    return p;
  };
  bf16* wqkv  = (bf16*)alloc((size_t)C3_ * C_ * 2);
  bf16* wproj = (bf16*)alloc((size_t)C_ * C_ * 2);
  bf16* wm1   = (bf16*)alloc((size_t)MLP_ * C_ * 2);
  bf16* wm2   = (bf16*)alloc((size_t)C_ * MLP_ * 2);
  // region1: q/kT/v (attention) overlapped with g (MLP hidden)
  const size_t szQ  = (size_t)NWH * SQ * HD_ * 2;    // 60,211,200
  const size_t szKT = (size_t)NWH * HD_ * KTPAD * 2; // 68,812,800
  const size_t szG  = (size_t)NTOK * MLP_ * 2;       // 201,326,592
  char* region1 = alloc(szG > (2 * szQ + szKT) ? szG : (2 * szQ + szKT));
  // region2: xw / attn_out / ln2-output, all [39200 or 32768, 768] bf16
  char* region2 = alloc((size_t)NTOKW * C_ * 2);
  bf16* qbuf   = (bf16*)region1;
  bf16* ktbuf  = (bf16*)(region1 + szQ);
  bf16* vbuf   = (bf16*)(region1 + szQ + szKT);
  bf16* gbuf   = (bf16*)region1;
  bf16* xwbuf  = (bf16*)region2;
  bf16* attnbf = (bf16*)region2;
  bf16* hbuf   = (bf16*)region2;

  // 1) fold/cast weights (7,077,888 elements = 27648 * 256 exactly)
  fold_weights_kernel<<<27648, 256, 0, stream>>>(
      qkv_w, lora_A, lora_B, proj_w, mlp_w1, mlp_w2, wqkv, wproj, wm1, wm2);

  // 2) LN1 + window partition
  ln1_window_kernel<<<NTOKW, 256, 0, stream>>>(x, norm1_w, norm1_b, xwbuf);

  // 3) QKV GEMM (mode 0)
  GemmParams p0 = {};
  p0.A = xwbuf; p0.W = wqkv; p0.M = NTOKW; p0.N = C3_; p0.K = C_;
  p0.mode = 0; p0.bias = qkv_b; p0.q = qbuf; p0.kT = ktbuf; p0.v = vbuf;
  gemm_bf16_kernel<<<dim3(C3_ / 128, (NTOKW + 63) / 64), 256, 0, stream>>>(p0);

  // 4) attention
  attention_kernel<<<NWH, 256, 0, stream>>>(qbuf, ktbuf, vbuf, rel_h, rel_w,
                                            attnbf);

  // 5) proj GEMM + unwindow + residual (mode 1)
  GemmParams p1 = {};
  p1.A = attnbf; p1.W = wproj; p1.M = NTOKW; p1.N = C_; p1.K = C_;
  p1.mode = 1; p1.bias = proj_b; p1.xres = x; p1.outf = out;
  gemm_bf16_kernel<<<dim3(C_ / 128, (NTOKW + 63) / 64), 256, 0, stream>>>(p1);

  // 6) LN2
  ln2_kernel<<<NTOK, 256, 0, stream>>>(out, norm2_w, norm2_b, hbuf);

  // 7) MLP up GEMM + GELU (mode 2)
  GemmParams p2 = {};
  p2.A = hbuf; p2.W = wm1; p2.M = NTOK; p2.N = MLP_; p2.K = C_;
  p2.mode = 2; p2.bias = mlp_b1; p2.gout = gbuf;
  gemm_bf16_kernel<<<dim3(MLP_ / 128, NTOK / 64), 256, 0, stream>>>(p2);

  // 8) MLP down GEMM + accumulate into d_out (mode 3)
  GemmParams p3 = {};
  p3.A = gbuf; p3.W = wm2; p3.M = NTOK; p3.N = C_; p3.K = MLP_;
  p3.mode = 3; p3.bias = mlp_b2; p3.outf = out;
  gemm_bf16_kernel<<<dim3(C_ / 128, NTOK / 64), 256, 0, stream>>>(p3);

  (void)in_sizes; (void)n_in; (void)out_size; (void)ws_size;
}
